// MGULayer_90580860272953
// MI455X (gfx1250) — matrix operand order repaired and checked by reading, not compiled, not run on hardware
//
#include <hip/hip_runtime.h>
#include <hip/hip_bf16.h>

typedef __attribute__((ext_vector_type(16))) _Float16 v16h;
typedef __attribute__((ext_vector_type(8)))  _Float16 v8h;
typedef __attribute__((ext_vector_type(8)))  float    v8f;

#define B_  64
#define T_  1024
#define I_  128
#define H_  256

// ---------------------------------------------------------------------------
// Packed B-tile layout (per 16x16x32 tile = 512 halves):
//   block0 (512B): element e=0..7  of lane L at tile*512 +       L*8 + e
//   block1 (512B): element e=8..15 of lane L at tile*512 + 256 + L*8 + (e-8)
// Semantic map (CDNA5 32x16 fp16 B layout, wave32):
//   n = nt*16 + (L & 15),  k = kt*32 + (L>>4)*8 + (e&7) + (e>>3)*16
// 16B/lane stride -> conflict-free ds_load_b128 pairs in LDS.
// ---------------------------------------------------------------------------
__global__ void mgu_pack_w(const float* __restrict__ W, _Float16* __restrict__ P, int K) {
    const int KT = K >> 5;
    const int total = KT * 16 * 512;
    for (int p = blockIdx.x * blockDim.x + threadIdx.x; p < total;
         p += gridDim.x * blockDim.x) {
        int tile = p >> 9;            // nt*KT + kt
        int r    = p & 511;
        int blk  = r >> 8;
        int lane = (r & 255) >> 3;
        int e    = blk * 8 + (r & 7);
        int nt = tile / KT;
        int kt = tile - nt * KT;
        int n = nt * 16 + (lane & 15);
        int k = kt * 32 + ((lane >> 4) << 3) + (e & 7) + ((e >> 3) << 4);
        P[p] = (_Float16)W[k * 256 + n];
    }
}

__device__ __forceinline__ v16h mgu_ldb(const _Float16* base, int tile, int lane) {
    const _Float16* tb = base + ((size_t)tile << 9) + (lane << 3);
    v8h lo = *(const v8h*)tb;
    v8h hi = *(const v8h*)(tb + 256);
    return __builtin_shufflevector(lo, hi, 0,1,2,3,4,5,6,7,8,9,10,11,12,13,14,15);
}

// A-fragment (16x32 fp16): p points at row_m_base + kt*32 + (lane>>4)*8.
__device__ __forceinline__ v16h mgu_lda(const _Float16* p) {
    v8h lo = *(const v8h*)p;
    v8h hi = *(const v8h*)(p + 16);
    return __builtin_shufflevector(lo, hi, 0,1,2,3,4,5,6,7,8,9,10,11,12,13,14,15);
}

#define WMMA_F16(A, Bv, C) \
    __builtin_amdgcn_wmma_f32_16x16x32_f16(false, (A), false, (Bv), (short)0, (C), false, false)

// ---------------------------------------------------------------------------
// Grid-parallel x-projection for layer 1: X = x @ Wx + b, fp16, [T][B][H].
// Block = 512 threads (16 waves); 32 A-rows staged in LDS with 4-dword skew.
// ---------------------------------------------------------------------------
template <int K, bool IN_F32, bool REMAP>
__global__ __launch_bounds__(512) void mgu_xproj(
    const void* __restrict__ in_v,
    const _Float16* __restrict__ Wfp, const _Float16* __restrict__ Wcp,
    const float* __restrict__ bf, const float* __restrict__ bc,
    _Float16* __restrict__ Xf, _Float16* __restrict__ Xc)
{
    constexpr int KT = K / 32;
    constexpr int SA = K + 8;
    __shared__ __attribute__((aligned(16))) _Float16 As[32 * SA];

    const int tid  = threadIdx.x;
    const int w    = tid >> 5;
    const int lane = tid & 31;
    const int Mbase = blockIdx.x * 32;

    for (int i = tid; i < 32 * K; i += 512) {
        int rr = i / K, k = i - rr * K;
        size_t src = (size_t)(Mbase + rr) * K + k;
        As[rr * SA + k] = IN_F32 ? (_Float16)(((const float*)in_v)[src])
                                 : ((const _Float16*)in_v)[src];
    }
    __syncthreads();

    const int mt  = w & 1;
    const int nt0 = w >> 1;               // pair: nt0, nt0+8
    v8f accf0 = {}, accf1 = {}, accc0 = {}, accc1 = {};
    const _Float16* arow = As + (mt * 16 + (lane & 15)) * SA + ((lane >> 4) << 3);

#pragma unroll
    for (int kt = 0; kt < KT; ++kt) {
        v16h a  = mgu_lda(arow + kt * 32);
        v16h f0 = mgu_ldb(Wfp, nt0 * KT + kt, lane);
        v16h f1 = mgu_ldb(Wfp, (nt0 + 8) * KT + kt, lane);
        v16h c0 = mgu_ldb(Wcp, nt0 * KT + kt, lane);
        v16h c1 = mgu_ldb(Wcp, (nt0 + 8) * KT + kt, lane);
        accf0 = WMMA_F16(a, f0, accf0);
        accf1 = WMMA_F16(a, f1, accf1);
        accc0 = WMMA_F16(a, c0, accc0);
        accc1 = WMMA_F16(a, c1, accc1);
    }

    const int khi8 = (lane >> 4) << 3;
#pragma unroll
    for (int half = 0; half < 2; ++half) {
        const int nc = (nt0 + half * 8) * 16 + (lane & 15);
        const float bfv = bf[nc], bcv = bc[nc];
        v8f af = half ? accf1 : accf0;
        v8f ac = half ? accc1 : accc0;
#pragma unroll
        for (int v = 0; v < 8; ++v) {
            const int r = Mbase + mt * 16 + khi8 + v;
            const size_t orow = REMAP ? ((size_t)(r & (T_ - 1)) * B_ + (r >> 10))
                                      : (size_t)r;
            Xf[orow * H_ + nc] = (_Float16)(af[v] + bfv);
            Xc[orow * H_ + nc] = (_Float16)(ac[v] + bcv);
        }
    }
}

// ===========================================================================
// 3-stage pipelined recurrent pass: one <<<3,1024>>> launch.
//   block0: layer-1 scan -> y1[t] (+flag1[t])
//   block1: per-step xproj of layer 2: y1[t] @ Wx2 + b -> X2[t] (+flag2[t])
//   block2: layer-2 scan (gated on flag2[t]) -> y2, hid2
// Wave w owns the (nt = w>>1) column pair of tiles mt0 = 2*(w&1), mt0+1 for
// BOTH gates, so each B fragment is loaded from LDS exactly once per wave.
// ===========================================================================
#define SMEM_HALVES (B_ * 264 + 2 * 16 * 8 * 512)   // 147968 halves = 289 KB

template <bool OUT_F32>
__device__ __forceinline__ void mgu_scan_body(
    const _Float16* __restrict__ Xf, const _Float16* __restrict__ Xc,
    const _Float16* __restrict__ WhfG, const _Float16* __restrict__ WhcG,
    void* __restrict__ y_out, float* __restrict__ hid_out,
    unsigned* waitFlag, unsigned* postFlag, _Float16* smem)
{
    constexpr int SH  = 264;
    constexpr int WSZ = 16 * 8 * 512;
    _Float16* hbuf = smem;
    _Float16* wfl  = smem + B_ * SH;
    _Float16* wcl  = wfl + WSZ;

    const int tid = threadIdx.x, w = tid >> 5, lane = tid & 31;
    {   // stage Wh tiles verbatim; zero h(0)
        const uint4* sf = (const uint4*)WhfG;
        const uint4* sc = (const uint4*)WhcG;
        uint4* df = (uint4*)wfl;
        uint4* dc = (uint4*)wcl;
        for (int i = tid; i < WSZ / 8; i += 1024) { df[i] = sf[i]; dc[i] = sc[i]; }
        for (int i = tid; i < B_ * H_; i += 1024)
            hbuf[(i >> 8) * SH + (i & 255)] = (_Float16)0.f;
    }
    __syncthreads();

    const int khi8 = (lane >> 4) << 3;
    const int nt   = w >> 1;
    const int mt0  = (w & 1) << 1;
    const int nc   = nt * 16 + (lane & 15);
    const int mb0  = mt0 * 16 + khi8;
    const _Float16* arow0 = hbuf + (mt0 * 16 + (lane & 15)) * SH + khi8;
    const _Float16* arow1 = arow0 + 16 * SH;

    for (int t = 0; t < T_; ++t) {
        if (waitFlag) {            // consumer: acquire this step's X tile
            if (tid == 0)
                while (__hip_atomic_load(waitFlag + t, __ATOMIC_ACQUIRE,
                                         __HIP_MEMORY_SCOPE_AGENT) == 0u)
                    __builtin_amdgcn_s_sleep(2);
            __syncthreads();
        } else if (t + 1 < T_) {   // free-running: prefetch next X block
            const _Float16* pb = (tid < 512) ? Xf : Xc;
            __builtin_prefetch(pb + (size_t)(t + 1) * (B_ * H_) + (size_t)(tid & 511) * 32, 0, 0);
        }

        v8f accf[2] = {}; v8f accc[2] = {};
#pragma unroll
        for (int kt = 0; kt < 8; ++kt) {   // B fragments loaded once per kt
            v16h bfv = mgu_ldb(wfl, nt * 8 + kt, lane);
            v16h bcv = mgu_ldb(wcl, nt * 8 + kt, lane);
            v16h a0  = mgu_lda(arow0 + kt * 32);
            v16h a1  = mgu_lda(arow1 + kt * 32);
            accf[0] = WMMA_F16(a0, bfv, accf[0]);
            accf[1] = WMMA_F16(a1, bfv, accf[1]);
            accc[0] = WMMA_F16(a0, bcv, accc[0]);
            accc[1] = WMMA_F16(a1, bcv, accc[1]);
        }

        float hold[2][8], xfv[2][8], xcv[2][8];
#pragma unroll
        for (int rep = 0; rep < 2; ++rep) {        // reads of h / X before barrier
            const int mb = mb0 + rep * 16;
            const size_t xbase = ((size_t)t * B_ + mb) * H_ + nc;
#pragma unroll
            for (int v = 0; v < 8; ++v) {
                hold[rep][v] = (float)hbuf[(mb + v) * SH + nc];
                xfv[rep][v]  = (float)Xf[xbase + (size_t)v * H_];
                xcv[rep][v]  = (float)Xc[xbase + (size_t)v * H_];
            }
        }
        __syncthreads();

#pragma unroll
        for (int rep = 0; rep < 2; ++rep) {
            const int mb = mb0 + rep * 16;
#pragma unroll
            for (int v = 0; v < 8; ++v) {
                float pf = accf[rep][v] + xfv[rep][v];   // bias folded into X
                float f  = 1.0f / (1.0f + __expf(-pf));
                float c  = tanhf(accc[rep][v] + xcv[rep][v]);
                float hn = f * hold[rep][v] + (1.0f - f) * c;
                const int row = mb + v;
                hbuf[row * SH + nc] = (_Float16)hn;
                if (OUT_F32)   // y2: [B][T][H] fp32 (reference layout)
                    ((float*)y_out)[((size_t)row * T_ + t) * H_ + nc] = hn;
                else           // y1: [T][B][H] fp16 (consumed per-step downstream)
                    ((_Float16*)y_out)[((size_t)t * B_ + row) * H_ + nc] = (_Float16)hn;
                if (t == T_ - 1) hid_out[row * H_ + nc] = hn;
            }
        }
        if (postFlag) {            // producer: publish y1[t]
            __threadfence();       // each thread's stores reach agent scope
            __syncthreads();
            if (tid == 0)
                __hip_atomic_store(postFlag + t, 1u, __ATOMIC_RELEASE,
                                   __HIP_MEMORY_SCOPE_AGENT);
        } else {
            __syncthreads();
        }
    }
}

__device__ __forceinline__ void mgu_xproj_step_body(
    const _Float16* __restrict__ y1,
    const _Float16* __restrict__ WxfG, const _Float16* __restrict__ WxcG,
    const float* __restrict__ bf, const float* __restrict__ bc,
    _Float16* __restrict__ X2f, _Float16* __restrict__ X2c,
    unsigned* flagIn, unsigned* flagOut, _Float16* smem)
{
    constexpr int SH  = 264;
    constexpr int WSZ = 16 * 8 * 512;
    _Float16* ybuf = smem;
    _Float16* wfl  = smem + B_ * SH;
    _Float16* wcl  = wfl + WSZ;

    const int tid = threadIdx.x, w = tid >> 5, lane = tid & 31;
    {
        const uint4* sf = (const uint4*)WxfG;
        const uint4* sc = (const uint4*)WxcG;
        uint4* df = (uint4*)wfl;
        uint4* dc = (uint4*)wcl;
        for (int i = tid; i < WSZ / 8; i += 1024) { df[i] = sf[i]; dc[i] = sc[i]; }
    }
    __syncthreads();

    const int khi8 = (lane >> 4) << 3;
    const int nt   = w >> 1;
    const int mt0  = (w & 1) << 1;
    const int nc   = nt * 16 + (lane & 15);
    const int mb0  = mt0 * 16 + khi8;
    const float bfv_ = bf[nc];
    const float bcv_ = bc[nc];
    const _Float16* arow0 = ybuf + (mt0 * 16 + (lane & 15)) * SH + khi8;
    const _Float16* arow1 = arow0 + 16 * SH;

    for (int t = 0; t < T_; ++t) {
        if (tid == 0)
            while (__hip_atomic_load(flagIn + t, __ATOMIC_ACQUIRE,
                                     __HIP_MEMORY_SCOPE_AGENT) == 0u)
                __builtin_amdgcn_s_sleep(2);
        __syncthreads();

        for (int i = tid; i < B_ * H_; i += 1024)     // stage y1[t] (32 KB)
            ybuf[(i >> 8) * SH + (i & 255)] = y1[(size_t)t * (B_ * H_) + i];
        __syncthreads();

        v8f accf[2] = {}; v8f accc[2] = {};
#pragma unroll
        for (int kt = 0; kt < 8; ++kt) {
            v16h bfv = mgu_ldb(wfl, nt * 8 + kt, lane);
            v16h bcv = mgu_ldb(wcl, nt * 8 + kt, lane);
            v16h a0  = mgu_lda(arow0 + kt * 32);
            v16h a1  = mgu_lda(arow1 + kt * 32);
            accf[0] = WMMA_F16(a0, bfv, accf[0]);
            accf[1] = WMMA_F16(a1, bfv, accf[1]);
            accc[0] = WMMA_F16(a0, bcv, accc[0]);
            accc[1] = WMMA_F16(a1, bcv, accc[1]);
        }

#pragma unroll
        for (int rep = 0; rep < 2; ++rep) {
            const size_t obase = ((size_t)t * B_ + mb0 + rep * 16) * H_ + nc;
#pragma unroll
            for (int v = 0; v < 8; ++v) {
                X2f[obase + (size_t)v * H_] = (_Float16)(accf[rep][v] + bfv_);
                X2c[obase + (size_t)v * H_] = (_Float16)(accc[rep][v] + bcv_);
            }
        }
        __threadfence();
        __syncthreads();
        if (tid == 0)
            __hip_atomic_store(flagOut + t, 1u, __ATOMIC_RELEASE,
                               __HIP_MEMORY_SCOPE_AGENT);
    }
}

__global__ __launch_bounds__(1024) void mgu_pipeline(
    const _Float16* X1f, const _Float16* X1c,
    const _Float16* wh1f, const _Float16* wh1c,
    const _Float16* wx2f, const _Float16* wx2c,
    const _Float16* wh2f, const _Float16* wh2c,
    const float* bf2, const float* bc2,
    _Float16* y1, _Float16* X2f, _Float16* X2c,
    float* y2, float* hid1, float* hid2,
    unsigned* flag1, unsigned* flag2)
{
    __shared__ __attribute__((aligned(16))) _Float16 smem[SMEM_HALVES];
    if (blockIdx.x == 0)
        mgu_scan_body<false>(X1f, X1c, wh1f, wh1c, (void*)y1, hid1,
                             nullptr, flag1, smem);
    else if (blockIdx.x == 1)
        mgu_xproj_step_body(y1, wx2f, wx2c, bf2, bc2, X2f, X2c,
                            flag1, flag2, smem);
    else
        mgu_scan_body<true>(X2f, X2c, wh2f, wh2c, (void*)y2, hid2,
                            flag2, nullptr, smem);
}

// ---------------------------------------------------------------------------
extern "C" void kernel_launch(void* const* d_in, const int* in_sizes, int n_in,
                              void* d_out, int out_size, void* d_ws, size_t ws_size,
                              hipStream_t stream) {
    const float* x   = (const float*)d_in[0];
    const float* Wf1 = (const float*)d_in[1];
    const float* bf1 = (const float*)d_in[2];
    const float* Wc1 = (const float*)d_in[3];
    const float* bc1 = (const float*)d_in[4];
    const float* Wf2 = (const float*)d_in[5];
    const float* bf2 = (const float*)d_in[6];
    const float* Wc2 = (const float*)d_in[7];
    const float* bc2 = (const float*)d_in[8];

    constexpr size_t WX1 = 4 * 16 * 512;
    constexpr size_t WH  = 8 * 16 * 512;
    constexpr size_t XSZ = (size_t)T_ * B_ * H_;

    unsigned* flag1 = (unsigned*)d_ws;
    unsigned* flag2 = flag1 + T_;
    _Float16* wx1f  = (_Float16*)(flag2 + T_);
    _Float16* wx1c  = wx1f + WX1;
    _Float16* wh1f  = wx1c + WX1;
    _Float16* wh1c  = wh1f + WH;
    _Float16* wx2f  = wh1c + WH;
    _Float16* wx2c  = wx2f + WH;
    _Float16* wh2f  = wx2c + WH;
    _Float16* wh2c  = wh2f + WH;
    _Float16* X1f   = wh2c + WH;
    _Float16* X1c   = X1f + XSZ;
    _Float16* X2f   = X1c + XSZ;
    _Float16* X2c   = X2f + XSZ;
    _Float16* y1    = X2c + XSZ;

    float* y2   = (float*)d_out;
    float* hid1 = y2 + (size_t)B_ * T_ * H_;
    float* hid2 = hid1 + (size_t)B_ * H_;

    hipMemsetAsync(flag1, 0, 2 * T_ * sizeof(unsigned), stream);

    // Split weights: x-part rows [0,K_in), h-part rows [K_in, K_in+256).
    mgu_pack_w<<<128, 256, 0, stream>>>(Wf1,            wx1f, 128);
    mgu_pack_w<<<128, 256, 0, stream>>>(Wc1,            wx1c, 128);
    mgu_pack_w<<<256, 256, 0, stream>>>(Wf1 + 128*256,  wh1f, 256);
    mgu_pack_w<<<256, 256, 0, stream>>>(Wc1 + 128*256,  wh1c, 256);
    mgu_pack_w<<<256, 256, 0, stream>>>(Wf2,            wx2f, 256);
    mgu_pack_w<<<256, 256, 0, stream>>>(Wc2,            wx2c, 256);
    mgu_pack_w<<<256, 256, 0, stream>>>(Wf2 + 256*256,  wh2f, 256);
    mgu_pack_w<<<256, 256, 0, stream>>>(Wc2 + 256*256,  wh2c, 256);

    // Layer-1 x-projection: whole-GPU parallel WMMA GEMM, [T][B][H] fp16 out.
    mgu_xproj<128, true, true><<<(B_*T_)/32, 512, 0, stream>>>(
        x, wx1f, wx1c, bf1, bc1, X1f, X1c);

    // Pipelined scans: scan1 -> per-step xproj2 -> scan2, one WGP each.
    mgu_pipeline<<<3, 1024, 0, stream>>>(
        X1f, X1c, wh1f, wh1c, wx2f, wx2c, wh2f, wh2c, bf2, bc2,
        y1, X2f, X2c, y2, hid1, hid2, flag1, flag2);
}
